// MyConv3d_package_84825604096303
// MI455X (gfx1250) — matrix operand order repaired and checked by reading, not compiled, not run on hardware
//
#include <hip/hip_runtime.h>
#include <hip/hip_bf16.h>

typedef _Float16 v16h __attribute__((ext_vector_type(16)));
typedef _Float16 v8h  __attribute__((ext_vector_type(8)));
typedef __fp16   h2raw __attribute__((ext_vector_type(2)));  // cvt_pkrtz result
typedef float    v8f  __attribute__((ext_vector_type(8)));

#define CIN   32
#define HDIM  32
#define WDIM  64
#define DDIM  64
#define COUT  64
#define X_CI  (HDIM * WDIM * DDIM)   /* 131072 */
#define X_H   (WDIM * DDIM)          /* 4096   */
#define X_B   (CIN * X_CI)           /* 4194304*/
#define O_CO  (HDIM * WDIM * DDIM)
#define O_B   (COUT * O_CO)
#define W_CO  864                    /* 32*27 weights per cout */
#define LDS_CO 872                   /* f16 units; 436 dwords -> bank-clean */
#define WELEMS (COUT * W_CO)         /* 55296 */
#define THREADS 512

/* Transposed X layout (f16): Xt[b][h][w][d][ci], strides in halves: */
#define XT_D  CIN                    /* 32     */
#define XT_W  (DDIM * CIN)           /* 2048   */
#define XT_HH (WDIM * DDIM * CIN)    /* 131072 */
#define XT_B  (HDIM * WDIM * DDIM * CIN) /* 4194304 */
#define XT_BYTES ((size_t)4 * XT_B * sizeof(_Float16)) /* 33554432 */

// Concatenate two 16B loads into one 16-half WMMA operand.
__device__ __forceinline__ v16h cat16(v8h lo, v8h hi) {
    return __builtin_shufflevector(lo, hi, 0, 1, 2, 3, 4, 5, 6, 7,
                                   8, 9, 10, 11, 12, 13, 14, 15);
}
__device__ __forceinline__ v16h ld_frag16(const _Float16* p) {
    return cat16(*(const v8h*)(p), *(const v8h*)(p + 16));
}

// ---------------------------------------------------------------------------
// Pre-pass: X (b,ci,h,w,d) f32  ->  Xt (b,h,w,d,ci) f16.
// Lanes along d: coalesced b32 reads; each thread writes its 32-ci run
// as four b128 stores.
// ---------------------------------------------------------------------------
extern "C" __global__ __launch_bounds__(256)
void transpose_x_f16(const float* __restrict__ X, _Float16* __restrict__ Xt) {
    int t = blockIdx.x * 256 + threadIdx.x;    // 524288 = 4*32*64*64
    int d = t & 63;
    int rest = t >> 6;
    int w = rest & 63;
    rest >>= 6;
    int h = rest & 31;
    int b = rest >> 5;

    const float* xp = X + b * X_B + h * X_H + w * DDIM + d;
    _Float16* op = Xt + (size_t)b * XT_B + h * XT_HH + w * XT_W + d * XT_D;

    float xs[32];
#pragma unroll
    for (int ci = 0; ci < 32; ++ci) xs[ci] = xp[ci * X_CI];  // imm offsets
#pragma unroll
    for (int k = 0; k < 4; ++k) {
        v8h v;
#pragma unroll
        for (int j = 0; j < 4; ++j) {
            h2raw p = __builtin_amdgcn_cvt_pkrtz(xs[k * 8 + 2 * j],
                                                 xs[k * 8 + 2 * j + 1]);
            v[2 * j]     = (_Float16)p.x;
            v[2 * j + 1] = (_Float16)p.y;
        }
        *(v8h*)(op + k * 8) = v;
    }
}

// ---------------------------------------------------------------------------
// Main kernel (transposed path): B fragment = two global_load_b128 of f16.
// Wave tile: 64 cout (4 M-tiles of 16) x 16 d positions; 27 taps x 4 WMMA.
// ---------------------------------------------------------------------------
extern "C" __global__ __launch_bounds__(THREADS)
void conv3d_wmma_f16_t(const _Float16* __restrict__ Xt,
                       const float* __restrict__ Wt,
                       const float* __restrict__ bias,
                       float* __restrict__ out) {
    extern __shared__ _Float16 wlds[];  // [co][tap][ci], co-stride LDS_CO

    for (int e = threadIdx.x; e < WELEMS; e += THREADS) {
        int co  = e / W_CO;
        int rem = e - co * W_CO;
        int ci  = rem / 27;
        int tap = rem - ci * 27;
        wlds[co * LDS_CO + tap * CIN + ci] = (_Float16)Wt[e];
    }
    __syncthreads();

    const int lane  = threadIdx.x & 31;
    const int wave  = threadIdx.x >> 5;
    const int n     = lane & 15;
    const int khalf = lane >> 4;

    int tile = blockIdx.x * (THREADS / 32) + wave;
    int b  = tile >> 13;
    int r  = tile & 8191;
    int h  = r >> 8;
    int r2 = r & 255;
    int w  = r2 >> 2;
    int d0 = (r2 & 3) << 4;

    v8f acc0 = {}, acc1 = {}, acc2 = {}, acc3 = {};

    // Per-lane base at kd=0: d index (d0+n-1), K-half selects ci run.
    const _Float16* xb = Xt + (size_t)b * XT_B + (d0 + n - 1) * XT_D + khalf * 8;
    const _Float16* wl = wlds + n * LDS_CO + khalf * 8;

    for (int kh = 0; kh < 3; ++kh) {
        int hh = h + kh - 1;
        bool hok = (unsigned)hh < (unsigned)HDIM;
        for (int kw = 0; kw < 3; ++kw) {
            int ww = w + kw - 1;
            bool hwok = hok && ((unsigned)ww < (unsigned)WDIM);
            const _Float16* xrow = xb + hh * XT_HH + ww * XT_W;
#pragma unroll
            for (int kd = 0; kd < 3; ++kd) {
                int dd = d0 + n + kd - 1;
                const int tap = kh * 9 + kw * 3 + kd;
                bool fast = hwok &&
                            ((unsigned)(d0 + kd - 1) <= (unsigned)(DDIM - 16));

                v8h lo, hi;
                if (fast) {                       // two b128, imm offsets
                    lo = *(const v8h*)(xrow + kd * XT_D);
                    hi = *(const v8h*)(xrow + kd * XT_D + 16);
                } else {
                    bool ok = hwok && ((unsigned)dd < (unsigned)DDIM);
                    lo = {}; hi = {};
                    if (ok) {
                        lo = *(const v8h*)(xrow + kd * XT_D);
                        hi = *(const v8h*)(xrow + kd * XT_D + 16);
                    }
                }
                v16h bf = cat16(lo, hi);          // K = 8*khalf + 0..7, 16 + 8*khalf + 0..7 = ci

                const _Float16* wt = wl + tap * CIN;
                v16h a0 = ld_frag16(wt);
                v16h a1 = ld_frag16(wt + 16 * LDS_CO);
                v16h a2 = ld_frag16(wt + 32 * LDS_CO);
                v16h a3 = ld_frag16(wt + 48 * LDS_CO);

                acc0 = __builtin_amdgcn_wmma_f32_16x16x32_f16(
                    false, a0, false, bf, (short)0, acc0, false, false);
                acc1 = __builtin_amdgcn_wmma_f32_16x16x32_f16(
                    false, a1, false, bf, (short)0, acc1, false, false);
                acc2 = __builtin_amdgcn_wmma_f32_16x16x32_f16(
                    false, a2, false, bf, (short)0, acc2, false, false);
                acc3 = __builtin_amdgcn_wmma_f32_16x16x32_f16(
                    false, a3, false, bf, (short)0, acc3, false, false);
            }
        }
    }

    int obase = b * O_B + h * X_H + w * DDIM + d0 + n;
    v8f accs[4] = {acc0, acc1, acc2, acc3};
#pragma unroll
    for (int m = 0; m < 4; ++m) {
        int cobase = m * 16 + khalf * 8;
#pragma unroll
        for (int rr = 0; rr < 8; ++rr) {
            int co = cobase + rr;
            out[obase + co * O_CO] = accs[m][rr] + bias[co];
        }
    }
}

// ---------------------------------------------------------------------------
// Fallback (workspace too small): direct fp32 gather + cvt_pk, same math.
// ---------------------------------------------------------------------------
extern "C" __global__ __launch_bounds__(THREADS)
void conv3d_wmma_f16(const float* __restrict__ X, const float* __restrict__ Wt,
                     const float* __restrict__ bias, float* __restrict__ out) {
    extern __shared__ _Float16 wlds[];

    for (int e = threadIdx.x; e < WELEMS; e += THREADS) {
        int co  = e / W_CO;
        int rem = e - co * W_CO;
        int ci  = rem / 27;
        int tap = rem - ci * 27;
        wlds[co * LDS_CO + tap * CIN + ci] = (_Float16)Wt[e];
    }
    __syncthreads();

    const int lane  = threadIdx.x & 31;
    const int wave  = threadIdx.x >> 5;
    const int n     = lane & 15;
    const int khalf = lane >> 4;

    int tile = blockIdx.x * (THREADS / 32) + wave;
    int b  = tile >> 13;
    int r  = tile & 8191;
    int h  = r >> 8;
    int r2 = r & 255;
    int w  = r2 >> 2;
    int d0 = (r2 & 3) << 4;

    v8f acc0 = {}, acc1 = {}, acc2 = {}, acc3 = {};

    const float* xb = X + b * X_B + khalf * 8 * X_CI;
    const _Float16* wl = wlds + n * LDS_CO + khalf * 8;

    for (int kh = 0; kh < 3; ++kh) {
        int hh = h + kh - 1;
        bool hok = (unsigned)hh < (unsigned)HDIM;
        for (int kw = 0; kw < 3; ++kw) {
            int ww = w + kw - 1;
            bool hwok = hok && ((unsigned)ww < (unsigned)WDIM);
            const float* xrow = xb + hh * X_H + ww * DDIM;
#pragma unroll
            for (int kd = 0; kd < 3; ++kd) {
                int dd = d0 + n + kd - 1;
                const int tap = kh * 9 + kw * 3 + kd;
                const float* xp = xrow + dd;
                bool fast = hwok &&
                            ((unsigned)(d0 + kd - 1) <= (unsigned)(DDIM - 16));

                float xs[16];
                if (fast) {
#pragma unroll
                    for (int j = 0; j < 16; ++j) xs[j] = xp[(j + (j & 8)) * X_CI];
                } else {
                    bool ok = hwok && ((unsigned)dd < (unsigned)DDIM);
#pragma unroll
                    for (int j = 0; j < 16; ++j) xs[j] = 0.0f;
                    if (ok) {
#pragma unroll
                        for (int j = 0; j < 16; ++j) xs[j] = xp[(j + (j & 8)) * X_CI];
                    }
                }
                v16h bf;
#pragma unroll
                for (int j = 0; j < 8; ++j) {
                    h2raw p = __builtin_amdgcn_cvt_pkrtz(xs[2 * j], xs[2 * j + 1]);
                    bf[2 * j]     = (_Float16)p.x;
                    bf[2 * j + 1] = (_Float16)p.y;
                }

                const _Float16* wt = wl + tap * CIN;
                v16h a0 = ld_frag16(wt);
                v16h a1 = ld_frag16(wt + 16 * LDS_CO);
                v16h a2 = ld_frag16(wt + 32 * LDS_CO);
                v16h a3 = ld_frag16(wt + 48 * LDS_CO);

                acc0 = __builtin_amdgcn_wmma_f32_16x16x32_f16(
                    false, a0, false, bf, (short)0, acc0, false, false);
                acc1 = __builtin_amdgcn_wmma_f32_16x16x32_f16(
                    false, a1, false, bf, (short)0, acc1, false, false);
                acc2 = __builtin_amdgcn_wmma_f32_16x16x32_f16(
                    false, a2, false, bf, (short)0, acc2, false, false);
                acc3 = __builtin_amdgcn_wmma_f32_16x16x32_f16(
                    false, a3, false, bf, (short)0, acc3, false, false);
            }
        }
    }

    int obase = b * O_B + h * X_H + w * DDIM + d0 + n;
    v8f accs[4] = {acc0, acc1, acc2, acc3};
#pragma unroll
    for (int m = 0; m < 4; ++m) {
        int cobase = m * 16 + khalf * 8;
#pragma unroll
        for (int rr = 0; rr < 8; ++rr) {
            int co = cobase + rr;
            out[obase + co * O_CO] = accs[m][rr] + bias[co];
        }
    }
}

extern "C" void kernel_launch(void* const* d_in, const int* in_sizes, int n_in,
                              void* d_out, int out_size, void* d_ws, size_t ws_size,
                              hipStream_t stream) {
    const float* X    = (const float*)d_in[0];
    const float* Wt   = (const float*)d_in[1];
    const float* bias = (const float*)d_in[2];
    float* out        = (float*)d_out;

    dim3 grid(2048);
    dim3 block(THREADS);
    size_t lds_bytes = (size_t)COUT * LDS_CO * sizeof(_Float16);  // 111616 B

    if (ws_size >= XT_BYTES && d_ws != nullptr) {
        _Float16* Xt = (_Float16*)d_ws;
        hipLaunchKernelGGL(transpose_x_f16, dim3(2048), dim3(256), 0, stream,
                           X, Xt);
        hipLaunchKernelGGL(conv3d_wmma_f16_t, grid, block, lds_bytes, stream,
                           Xt, Wt, bias, out);
    } else {
        hipLaunchKernelGGL(conv3d_wmma_f16, grid, block, lds_bytes, stream,
                           X, Wt, bias, out);
    }
}
